// GeneralCRF_50233937494628
// MI455X (gfx1250) — matrix operand order repaired and checked by reading, not compiled, not run on hardware
//
#include <hip/hip_runtime.h>
#include <hip/hip_bf16.h>

typedef __attribute__((ext_vector_type(16))) _Float16 v16h;
typedef __attribute__((ext_vector_type(8)))  float    v8f;

#define B_  8
#define N_  1024
#define S_  512
#define E_  256
#define F_  256
#define D_  32
#define K_  64   // BEAM

// ---------------- Kernel 1: beam top-64 selection + unary gold term --------
__global__ __launch_bounds__(256) void k_beam(
    const float* __restrict__ unaries, const int* __restrict__ masks,
    const int* __restrict__ targets, int* __restrict__ beam_targets,
    float* __restrict__ gold_unary)
{
  __shared__ float sel[8][S_];
  __shared__ float bval[8][K_];
  const int tid  = threadIdx.x;
  const int w    = tid >> 5;
  const int lane = tid & 31;
  const int row  = blockIdx.x * 8 + w;          // row in [0, B*N)
  const float mf = masks[row] ? 1.0f : 0.0f;
  const int tgt  = targets[row];
  const float* up = unaries + (size_t)row * S_;

  for (int s = lane; s < S_; s += 32) {
    float u = up[s] * mf;
    sel[w][s] = (s == tgt) ? 1e30f : u;         // force target to rank first
  }
  // 64 rounds of wave-wide argmax (tie -> smaller index, matching lax.top_k)
  for (int k = 0; k < K_; ++k) {
    float best = -3.4e38f; int bidx = 1 << 30;
    for (int s = lane; s < S_; s += 32) {
      float v = sel[w][s];
      if (v > best || (v == best && s < bidx)) { best = v; bidx = s; }
    }
    for (int off = 16; off > 0; off >>= 1) {
      float ov = __shfl_xor(best, off, 32);
      int   oi = __shfl_xor(bidx, off, 32);
      if (ov > best || (ov == best && oi < bidx)) { best = ov; bidx = oi; }
    }
    if (lane == 0) {
      sel[w][bidx] = -3.0e38f;                  // remove from candidates
      beam_targets[(size_t)row * K_ + k] = bidx;
      bval[w][k] = up[bidx] * mf;               // original (un-spiked) value
    }
  }
  // log_softmax(beam_unary)[0] = bval[0] - logsumexp(bval)
  float v0 = bval[w][lane];
  float v1 = bval[w][lane + 32];
  float m = fmaxf(v0, v1);
  for (int off = 16; off > 0; off >>= 1) m = fmaxf(m, __shfl_xor(m, off, 32));
  float sum = expf(v0 - m) + expf(v1 - m);
  for (int off = 16; off > 0; off >>= 1) sum += __shfl_xor(sum, off, 32);
  if (lane == 0) {
    float lse = m + logf(sum);
    gold_unary[row] = (mf != 0.0f) ? (bval[w][0] - lse) : 0.0f;
  }
}

// ------------- Kernel 2: ew = relu(feat @ W + b), f16 out, WMMA ------------
// M = B*E = 2048 (edge rows), K = 2F = 512, Nout = D*D = 1024.
// W tile is staged into LDS pre-swizzled into WMMA-B fragment layout so each
// lane fetches its whole 32-byte fragment with vector ds loads.
__global__ __launch_bounds__(128) void k_gemm_ew(
    const float* __restrict__ nf, const int* __restrict__ edges,
    const float* __restrict__ Wd, const float* __restrict__ bd,
    _Float16* __restrict__ ewh)
{
  __shared__ __align__(32) _Float16 bfrag[4][32][16];   // [nt][lane][elem]
  const int tid  = threadIdx.x;
  const int w    = tid >> 5;
  const int lane = tid & 31;
  const int hf   = lane >> 4;
  const int l16  = lane & 15;
  const int mbase = blockIdx.x * 64 + w * 16;   // wave's 16-row strip
  const int n0    = blockIdx.y * 64;

  const int r  = mbase + l16;                   // A-row this lane supplies
  const int b  = r >> 8;                        // / E_
  const int e0 = edges[r * 2 + 0];
  const int e1 = edges[r * 2 + 1];
  const float* p0 = nf + ((size_t)b * N_ + e0) * F_;
  const float* p1 = nf + ((size_t)b * N_ + e1) * F_;

  v8f acc[4] = {};
  for (int k0 = 0; k0 < 2 * F_; k0 += 32) {
    // stage 32x64 W tile: coalesced global reads, swizzled LDS writes.
    // dest fragment: bfrag[nt][hf*16+l16][i] = W[k0+16*hf+i][n0+nt*16+l16]
    for (int idx = tid; idx < 2048; idx += 128) {
      int kk = idx >> 6, nn = idx & 63;
      float v = Wd[(size_t)(k0 + kk) * 1024 + n0 + nn];
      bfrag[nn >> 4][((kk >> 4) << 4) | (nn & 15)][kk & 15] = (_Float16)v;
    }
    __builtin_prefetch(&Wd[(size_t)(k0 + 32) * 1024 + n0], 0, 0);
    __syncthreads();

    // A operand: feat = concat(f0, f1); two aligned 8-float chunks per lane
    int Ka = k0 + 16 * hf;
    int Kb = k0 + 16 * hf + 8;
    const float* qa = (Ka < F_) ? (p0 + Ka) : (p1 + (Ka - F_));
    const float* qb = (Kb < F_) ? (p0 + Kb) : (p1 + (Kb - F_));
    float4 c0 = *(const float4*)(qa);
    float4 c1 = *(const float4*)(qa + 4);
    float4 c2 = *(const float4*)(qb);
    float4 c3 = *(const float4*)(qb + 4);
    v16h a;
    a[0]=(_Float16)c0.x;  a[1]=(_Float16)c0.y;  a[2]=(_Float16)c0.z;  a[3]=(_Float16)c0.w;
    a[4]=(_Float16)c1.x;  a[5]=(_Float16)c1.y;  a[6]=(_Float16)c1.z;  a[7]=(_Float16)c1.w;
    a[8]=(_Float16)c2.x;  a[9]=(_Float16)c2.y;  a[10]=(_Float16)c2.z; a[11]=(_Float16)c2.w;
    a[12]=(_Float16)c3.x; a[13]=(_Float16)c3.y; a[14]=(_Float16)c3.z; a[15]=(_Float16)c3.w;

#pragma unroll
    for (int nt = 0; nt < 4; ++nt) {
      v16h bm = *(const v16h*)(&bfrag[nt][lane][0]);   // 2x ds_load_b128
      acc[nt] = __builtin_amdgcn_wmma_f32_16x16x32_f16(
          false, a, false, bm, (short)0, acc[nt], false, false);
    }
    __syncthreads();
  }
  // bias + relu, f16 store
  for (int nt = 0; nt < 4; ++nt) {
    int col = n0 + nt * 16 + l16;
    float bias = bd[col];
#pragma unroll
    for (int v = 0; v < 8; ++v) {
      int rowg = mbase + v + 8 * hf;               // C: M = v + 8*half
      float x = acc[nt][v] + bias;
      x = x > 0.0f ? x : 0.0f;
      ewh[(size_t)rowg * 1024 + col] = (_Float16)x;
    }
  }
}

// ------- Kernel 3: per-edge  P = s0 @ ew @ s1^T  + log-softmax gold -------
// All WMMA operands staged in LDS pre-swizzled to fragment layout; s1 stays
// row-major because its B fragment is naturally 32 contiguous bytes.
__global__ __launch_bounds__(128) void k_edge(
    const float* __restrict__ bin_embed, const int* __restrict__ edges,
    const int* __restrict__ bmask, const int* __restrict__ beam_targets,
    const _Float16* __restrict__ ewh, float* __restrict__ gold_bin)
{
  __shared__ __align__(32) _Float16 s0frag[4][32][16];  // A frags, step 1
  __shared__ __align__(32) _Float16 s1h[64 * 32];       // row-major [l][j]
  __shared__ __align__(32) _Float16 ewfrag[2][32][16];  // B frags, step 1
  __shared__ __align__(32) _Float16 thfrag[4][32][16];  // A frags, step 2
  __shared__ float red[128];
  __shared__ float sP00;

  const int r    = blockIdx.x;                  // edge row in [0, B*E)
  const int b    = r >> 8;
  const int tid  = threadIdx.x;
  const int w    = tid >> 5;
  const int lane = tid & 31;
  const int hf   = lane >> 4;
  const int l16  = lane & 15;
  const int e0   = edges[r * 2 + 0];
  const int e1   = edges[r * 2 + 1];
  const int* bt0 = beam_targets + ((size_t)b * N_ + e0) * K_;
  const int* bt1 = beam_targets + ((size_t)b * N_ + e1) * K_;
  const float* emb0 = bin_embed;                // [S, D]
  const float* emb1 = bin_embed + S_ * D_;

  // stage s0 (swizzled to A-fragment layout) and s1 (row-major)
  for (int idx = tid; idx < 2048; idx += 128) {
    int k = idx >> 5, j = idx & 31;
    _Float16 v0 = (_Float16)emb0[(size_t)bt0[k] * D_ + j];
    int wv = k >> 4, m = k & 15;
    int g = j >> 4, rem = j & 15;
    s0frag[wv][((j >> 4) << 4) | m][j & 15] = v0;
    s1h[idx] = (_Float16)emb1[(size_t)bt1[k] * D_ + j];
  }
  // stage ew (swizzled to B-fragment layout): ewfrag[nt][hf*16+l][i] = ew[16*hf+i][nt*16+l]
  for (int idx = tid; idx < 1024; idx += 128) {
    int kk = idx >> 5, nn = idx & 31;
    _Float16 v = ewh[(size_t)r * 1024 + idx];
    ewfrag[nn >> 4][((kk >> 4) << 4) | (nn & 15)][kk & 15] = v;
  }
  __syncthreads();

  // ---- step 1: T(16x32 per wave) = s0_strip(16x32) @ ew(32x32) ----
  v16h a = *(const v16h*)(&s0frag[w][lane][0]);
  v8f t0, t1;
  {
    v16h bm = *(const v16h*)(&ewfrag[0][lane][0]);
    v8f z = {};
    t0 = __builtin_amdgcn_wmma_f32_16x16x32_f16(false, a, false, bm, (short)0, z, false, false);
  }
  {
    v16h bm = *(const v16h*)(&ewfrag[1][lane][0]);
    v8f z = {};
    t1 = __builtin_amdgcn_wmma_f32_16x16x32_f16(false, a, false, bm, (short)0, z, false, false);
  }
  // write T directly into step-2 A-fragment layout:
  // value (m = v+8*hf, n) -> thfrag[w][16*(n>>4) + m][n&15]
#pragma unroll
  for (int v = 0; v < 8; ++v) {
    int m = v + 8 * hf;
    int lanep = m;
    thfrag[w][lanep][l16]      = (_Float16)t0[v];   // n = l16
    thfrag[w][16 + lanep][l16] = (_Float16)t1[v];   // n = 16 + l16
  }
  __syncthreads();

  // ---- step 2: P strip (16x64) = T(16x32) @ s1^T(32x64) ----
  v16h a2 = *(const v16h*)(&thfrag[w][lane][0]);
  v8f p[4];
#pragma unroll
  for (int nt = 0; nt < 4; ++nt) {
    // B[K=j][N=l] = s1[l][j]; fragment = s1h[(nt*16+l16)*32 + 16*hf .. +15]
    v16h bm = *(const v16h*)(&s1h[(nt * 16 + l16) * 32 + 16 * hf]);
    v8f z = {};
    p[nt] = __builtin_amdgcn_wmma_f32_16x16x32_f16(false, a2, false, bm, (short)0, z, false, false);
  }
  if (tid == 0) sP00 = p[0][0];                 // element (k=0, l=0)

  // ---- block-wide logsumexp over 4096 values (32 per thread) ----
  float mx = -3.4e38f;
  for (int nt = 0; nt < 4; ++nt)
#pragma unroll
    for (int v = 0; v < 8; ++v) mx = fmaxf(mx, p[nt][v]);
  red[tid] = mx;
  __syncthreads();
  for (int s = 64; s > 0; s >>= 1) {
    if (tid < s) red[tid] = fmaxf(red[tid], red[tid + s]);
    __syncthreads();
  }
  float gmax = red[0];
  __syncthreads();
  float sm = 0.0f;
  for (int nt = 0; nt < 4; ++nt)
#pragma unroll
    for (int v = 0; v < 8; ++v) sm += expf(p[nt][v] - gmax);
  red[tid] = sm;
  __syncthreads();
  for (int s = 64; s > 0; s >>= 1) {
    if (tid < s) red[tid] += red[tid + s];
    __syncthreads();
  }
  if (tid == 0) {
    float lse = gmax + logf(red[0]);
    gold_bin[r] = bmask[r] ? (sP00 - lse) : 0.0f;
  }
}

// ---------------- Kernel 4: final deterministic reduction -> NLL -----------
__global__ __launch_bounds__(256) void k_final(
    const float* __restrict__ gold_unary, const float* __restrict__ gold_bin,
    const int* __restrict__ masks, float* __restrict__ out)
{
  __shared__ float red[256];
  const int tid = threadIdx.x;
  float acc = 0.0f;
  for (int b = 0; b < B_; ++b) {
    float su = 0.0f, sb = 0.0f, sm = 0.0f;
    for (int i = tid; i < N_; i += 256) su += gold_unary[b * N_ + i];
    for (int i = tid; i < E_; i += 256) sb += gold_bin[b * E_ + i];
    for (int i = tid; i < N_; i += 256) sm += masks[b * N_ + i] ? 1.0f : 0.0f;

    red[tid] = su; __syncthreads();
    for (int s = 128; s > 0; s >>= 1) { if (tid < s) red[tid] += red[tid + s]; __syncthreads(); }
    float SU = red[0]; __syncthreads();
    red[tid] = sb; __syncthreads();
    for (int s = 128; s > 0; s >>= 1) { if (tid < s) red[tid] += red[tid + s]; __syncthreads(); }
    float SB = red[0]; __syncthreads();
    red[tid] = sm; __syncthreads();
    for (int s = 128; s > 0; s >>= 1) { if (tid < s) red[tid] += red[tid + s]; __syncthreads(); }
    float SM = red[0]; __syncthreads();
    acc += (SU + SB) / SM;
  }
  if (tid == 0) out[0] = -acc / (float)B_;
}

extern "C" void kernel_launch(void* const* d_in, const int* in_sizes, int n_in,
                              void* d_out, int out_size, void* d_ws, size_t ws_size,
                              hipStream_t stream)
{
  const float* unaries   = (const float*)d_in[0];
  const int*   masks     = (const int*)d_in[1];
  const int*   edges     = (const int*)d_in[2];
  const int*   bmask     = (const int*)d_in[3];
  const float* nf        = (const float*)d_in[4];
  const int*   targets   = (const int*)d_in[5];
  const float* bin_embed = (const float*)d_in[6];
  const float* Wd        = (const float*)d_in[7];
  const float* bd        = (const float*)d_in[8];

  char* ws = (char*)d_ws;
  int*      beam_targets = (int*)(ws);                          // 2 MB
  _Float16* ewh          = (_Float16*)(ws + (2u << 20));        // 4 MB
  float*    gold_unary   = (float*)(ws + (6u << 20));           // 32 KB
  float*    gold_bin     = (float*)(ws + (6u << 20) + 32768);   // 8 KB
  float*    out          = (float*)d_out;

  k_beam<<<(B_ * N_) / 8, 256, 0, stream>>>(unaries, masks, targets,
                                            beam_targets, gold_unary);
  dim3 g2(32, 16);  // (2048/64) x (1024/64) tiles
  k_gemm_ew<<<g2, 128, 0, stream>>>(nf, edges, Wd, bd, ewh);
  k_edge<<<B_ * E_, 128, 0, stream>>>(bin_embed, edges, bmask,
                                      beam_targets, ewh, gold_bin);
  k_final<<<1, 256, 0, stream>>>(gold_unary, gold_bin, masks, out);
}
